// RecurrentMambaCellRef_27255862460671
// MI455X (gfx1250) — matrix operand order repaired and checked by reading, not compiled, not run on hardware
//
#include <hip/hip_runtime.h>
#include <hip/hip_bf16.h>
#include <cstdint>
#include <cstddef>

// ---------------- CDNA5 (gfx1250) types ----------------
typedef __bf16 bf16_t;
typedef __attribute__((ext_vector_type(16))) __bf16 v16bf;
typedef __attribute__((ext_vector_type(8)))  float  v8f;
typedef unsigned int v4u __attribute__((ext_vector_type(4)));
typedef int v4i __attribute__((ext_vector_type(4)));
typedef int v8i __attribute__((ext_vector_type(8)));

union frag16 { uint4 u[2]; v16bf v; };

__device__ __forceinline__ v8f zero8() {
  v8f r;
#pragma unroll
  for (int i = 0; i < 8; ++i) r[i] = 0.f;
  return r;
}

__device__ __forceinline__ v8f wmma_bf16(v16bf a, v16bf b, v8f c) {
  // v_wmma_f32_16x16x32_bf16  (neg_a, A, neg_b, B, c_mod, C, reuse_a, reuse_b)
  return __builtin_amdgcn_wmma_f32_16x16x32_bf16(false, a, false, b, (short)0, c,
                                                 false, false);
}

__device__ __forceinline__ float siluf(float v) {
  return v * (1.f / (1.f + __expf(-v)));
}

// ---------------- problem constants ----------------
#define DMODEL  1024
#define DSTATE  16
#define DINNER  2048
#define HEADDIM 64
#define NHEADS  32
#define CONVDIM 2080
#define NCOLS   2112          // CONVDIM + NHEADS (xBC + dt columns of in_proj)
#define BSZ     8
#define TSEQ    2049
#define TPAD    2080          // 65 * 32, K padding for WMMA
#define MROWS   (BSZ * TSEQ)  // 16392

// GEMM tiling
#define BM   128
#define BN   64
#define BK   32
#define ASTR 40   // LDS row stride (elems): 80B rows, matches TDM pad (16dw + 4dw)
#define BSTR 40

#if __has_builtin(__builtin_amdgcn_tensor_load_to_lds)
#define USE_TDM 1
#else
#define USE_TDM 0
#endif

#if USE_TDM
// ---- Tensor Data Mover: 2-D tile (tile_d0 x tile_d1 elems, bf16) -> LDS ----
// D# per CDNA5 ISA ch.8. pad_interval=3 (every 16 DWORDs = one 32-elem row),
// pad_amount=3 (4 DWORDs = 8 elems) -> LDS row stride 40 elems = ASTR/BSTR.
__device__ __forceinline__ void tdm_load_2d(unsigned lds_addr,
                                            unsigned long long gaddr,
                                            unsigned tensor_d0, unsigned tensor_d1,
                                            unsigned tile_d0, unsigned tile_d1,
                                            unsigned stride0) {
  v4u g0;
  g0.x = 1u;                                   // count=1 (valid user descriptor)
  g0.y = lds_addr;                             // LDS byte address
  g0.z = (unsigned)(gaddr & 0xFFFFFFFFull);    // global_addr[31:0]
  g0.w = (unsigned)((gaddr >> 32) & 0x01FFFFFFull) | (2u << 30);  // [56:32] | type=2
  v8i g1;
  g1[0] = (int)((1u << 16)        // data_size = 1 -> 2 bytes (bf16)
              | (1u << 20)        // pad_enable
              | (3u << 22)        // pad_interval: 16 DWORDs
              | (3u << 25));      // pad_amount: 4 DWORDs
  g1[1] = (int)((tensor_d0 & 0xFFFFu) << 16);                         // dim0[15:0]
  g1[2] = (int)(((tensor_d0 >> 16) & 0xFFFFu) | ((tensor_d1 & 0xFFFFu) << 16));
  g1[3] = (int)(((tensor_d1 >> 16) & 0xFFFFu) | ((tile_d0 & 0xFFFFu) << 16));
  g1[4] = (int)(tile_d1 & 0xFFFFu);            // tile_dim1 (tile_dim2 = 0)
  g1[5] = (int)stride0;                        // tensor_dim0_stride[31:0]
  g1[6] = 0;                                   // stride0[47:32] | stride1[15:0]
  g1[7] = 0;                                   // stride1[47:16]
  v4i z4 = {0, 0, 0, 0};
#if defined(__clang_major__) && (__clang_major__ >= 23)
  v8i z8 = {0, 0, 0, 0, 0, 0, 0, 0};
  __builtin_amdgcn_tensor_load_to_lds(g0, g1, z4, z4, z8, 0);
#else
  __builtin_amdgcn_tensor_load_to_lds(g0, g1, z4, z4, 0);
#endif
}
#endif  // USE_TDM

// ---------------- cast kernels ----------------
__global__ __launch_bounds__(256) void k_cast_seq(const float* __restrict__ x_t,
                                                  const float* __restrict__ history,
                                                  bf16_t* __restrict__ seqb) {
  size_t idx = (size_t)blockIdx.x * 256 + threadIdx.x;   // MROWS*DMODEL
  int r = (int)(idx >> 10);
  int d = (int)(idx & 1023);
  int b = r / TSEQ;
  int t = r - b * TSEQ;
  float v = (t < TSEQ - 1) ? history[((size_t)b * (TSEQ - 1) + t) * DMODEL + d]
                           : x_t[(size_t)b * DMODEL + d];
  seqb[idx] = (bf16_t)v;
}

__global__ __launch_bounds__(256) void k_cast_w(const float* __restrict__ in_proj_w,
                                                bf16_t* __restrict__ w2b) {
  size_t idx = (size_t)blockIdx.x * 256 + threadIdx.x;   // NCOLS*DMODEL
  int e = (int)(idx >> 10);
  int k = (int)(idx & 1023);
  w2b[idx] = (bf16_t)in_proj_w[(size_t)(DINNER + e) * DMODEL + k];
}

// ---------------- main in-projection GEMM (TDM + WMMA bf16 -> f32) ----------------
// C[MROWS][NCOLS] = seqb[MROWS][1024] * w2b[NCOLS][1024]^T
__global__ __launch_bounds__(256) void k_gemm1(const bf16_t* __restrict__ A,
                                               const bf16_t* __restrict__ Bm,
                                               float* __restrict__ C) {
  __shared__ __align__(16) bf16_t Atile[BM * ASTR];
  __shared__ __align__(16) bf16_t Btile[BN * BSTR];

  const int tid      = threadIdx.x;
  const int rowbase  = blockIdx.x * BM;
  const int colbase  = blockIdx.y * BN;
  const int lane     = tid & 31;
  const int wv       = tid >> 5;
  const int wm       = (wv >> 1) * 32;   // 0,32,64,96
  const int wn       = (wv & 1) * 32;    // 0,32
  const int khalf    = lane >> 4;
  const int l15      = lane & 15;

  v8f acc[2][2];
#pragma unroll
  for (int mt = 0; mt < 2; ++mt)
#pragma unroll
    for (int nt = 0; nt < 2; ++nt) acc[mt][nt] = zero8();

  for (int k0 = 0; k0 < DMODEL; k0 += BK) {
#if USE_TDM
    // Leader wave drives the Tensor Data Mover; TDM OOB returns zeros, so the
    // partial last M-block needs no guards. Padding gives the 80B LDS rows.
    if (tid < 32) {
      tdm_load_2d((unsigned)(uintptr_t)(void*)Atile,
                  (unsigned long long)(uintptr_t)A + ((size_t)rowbase * DMODEL + k0) * 2,
                  DMODEL, MROWS, BK, BM, DMODEL);
      tdm_load_2d((unsigned)(uintptr_t)(void*)Btile,
                  (unsigned long long)(uintptr_t)Bm + ((size_t)colbase * DMODEL + k0) * 2,
                  DMODEL, NCOLS, BK, BN, DMODEL);
      __builtin_amdgcn_s_wait_tensorcnt(0);
    }
    // Warm GL2 for the next K-slab while the TDM streams this one.
    {
      int gr = rowbase + (tid >> 1);
      if (gr < MROWS && k0 + BK < DMODEL)
        __builtin_prefetch((const void*)(A + (size_t)gr * DMODEL + k0 + BK + (tid & 1) * 16),
                           0, 1);
    }
#else
    // fallback: manual staging, 2x uint4 per thread for A, 1x for B
#pragma unroll
    for (int i = 0; i < 2; ++i) {
      int li = tid + i * 256;
      int r  = li >> 2;
      int c8 = (li & 3) * 8;
      int gr = rowbase + r;
      uint4 v = {0u, 0u, 0u, 0u};
      if (gr < MROWS) {
        v = *(const uint4*)(A + (size_t)gr * DMODEL + k0 + c8);
        if (k0 + BK < DMODEL)
          __builtin_prefetch((const void*)(A + (size_t)gr * DMODEL + k0 + BK + c8), 0, 1);
      }
      *(uint4*)(Atile + r * ASTR + c8) = v;
    }
    {
      int r  = tid >> 2;
      int c8 = (tid & 3) * 8;
      uint4 v = *(const uint4*)(Bm + (size_t)(colbase + r) * DMODEL + k0 + c8);
      *(uint4*)(Btile + r * BSTR + c8) = v;
    }
#endif
    __syncthreads();

    frag16 afr[2], bfr[2];
#pragma unroll
    for (int mt = 0; mt < 2; ++mt) {   // ISA 16-bit A 16x32 layout
      const bf16_t* ap = Atile + (wm + mt * 16 + l15) * ASTR + khalf * 8;
      afr[mt].u[0] = *(const uint4*)(ap);
      afr[mt].u[1] = *(const uint4*)(ap + 16);
    }
#pragma unroll
    for (int nt = 0; nt < 2; ++nt) {   // B 32x16: lane=N, K in ISA 16-bit table order
      const bf16_t* bp = Btile + (wn + nt * 16 + l15) * BSTR + khalf * 8;
      bfr[nt].u[0] = *(const uint4*)(bp);
      bfr[nt].u[1] = *(const uint4*)(bp + 16);
    }
#pragma unroll
    for (int mt = 0; mt < 2; ++mt)
#pragma unroll
      for (int nt = 0; nt < 2; ++nt)
        acc[mt][nt] = wmma_bf16(afr[mt].v, bfr[nt].v, acc[mt][nt]);
    __syncthreads();
  }

#pragma unroll
  for (int mt = 0; mt < 2; ++mt)
#pragma unroll
    for (int nt = 0; nt < 2; ++nt) {
      int gc = colbase + wn + nt * 16 + l15;
#pragma unroll
      for (int j = 0; j < 8; ++j) {
        int gr = rowbase + wm + mt * 16 + 8 * khalf + j;
        if (gr < MROWS) C[(size_t)gr * NCOLS + gc] = acc[mt][nt][j];
      }
    }
}

// ---------------- dt: softplus + cumsum + decay weights ----------------
// one wave per (b,h); wts[b,h,t] = dt_t * exp(A_h * (S_T - S_t))
__global__ __launch_bounds__(32) void k_dt(const float* __restrict__ xbcdt,
                                           const float* __restrict__ dt_bias,
                                           const float* __restrict__ A_log,
                                           float* __restrict__ wts) {
  int bh = blockIdx.x;
  int b = bh >> 5, h = bh & 31;
  int lane = threadIdx.x;
  float bias = dt_bias[h];
  float Ah   = -__expf(A_log[h]);
  float* wrow = wts + (size_t)bh * TPAD;

  // pass 1: dt -> wrow, total sum
  float total = 0.f;
  for (int t0 = 0; t0 < TSEQ; t0 += 32) {
    int t = t0 + lane;
    float dt = 0.f;
    if (t < TSEQ) {
      float v = xbcdt[(size_t)(b * TSEQ + t) * NCOLS + CONVDIM + h] + bias;
      dt = (v > 20.f) ? v : log1pf(__expf(v));
      wrow[t] = dt;
    }
    float s = dt;
#pragma unroll
    for (int d = 1; d < 32; d <<= 1) s += __shfl_xor(s, d);
    total += s;
  }
  // pass 2: inclusive prefix scan -> weights
  float pcarry = 0.f;
  for (int t0 = 0; t0 < TSEQ; t0 += 32) {
    int t = t0 + lane;
    float dt = (t < TSEQ) ? wrow[t] : 0.f;
    float s = dt;
#pragma unroll
    for (int d = 1; d < 32; d <<= 1) {
      float u = __shfl_up(s, d);
      if (lane >= d) s += u;
    }
    float S = pcarry + s;
    if (t < TSEQ) wrow[t] = dt * __expf(Ah * (total - S));
    pcarry += __shfl(s, 31);
  }
}

// ---------------- causal conv + SiLU + weight, x channels ----------------
// xw[b,h,p,t] = silu(conv(x)) * wts[b,h,t]  (bf16, t padded with zeros)
__global__ __launch_bounds__(256) void k_convx(const float* __restrict__ xbcdt,
                                               const float* __restrict__ conv_w,
                                               const float* __restrict__ conv_b,
                                               const float* __restrict__ wts,
                                               bf16_t* __restrict__ xw,
                                               float* __restrict__ xlast) {
  int idx = blockIdx.x * 256 + threadIdx.x;  // 8*2048
  int b = idx >> 11;
  int c = idx & 2047;
  int h = c >> 6, p = c & 63;
  float w0 = conv_w[c * 4 + 0], w1 = conv_w[c * 4 + 1];
  float w2 = conv_w[c * 4 + 2], w3 = conv_w[c * 4 + 3];
  float bias = conv_b[c];
  const float* col  = xbcdt + (size_t)b * TSEQ * NCOLS + c;
  const float* wrow = wts + (size_t)(b * NHEADS + h) * TPAD;
  bf16_t* orow = xw + (size_t)((b * NHEADS + h) * HEADDIM + p) * TPAD;

  float x0 = 0.f, x1 = 0.f, x2 = 0.f;
  for (int t = 0; t < TSEQ; ++t) {
    float x3 = col[(size_t)t * NCOLS];
    float s  = siluf(x0 * w0 + x1 * w1 + x2 * w2 + x3 * w3 + bias);
    orow[t]  = (bf16_t)(s * wrow[t]);
    if (t == TSEQ - 1) xlast[(b * NHEADS + h) * HEADDIM + p] = s;
    x0 = x1; x1 = x2; x2 = x3;
  }
  for (int t = TSEQ; t < TPAD; ++t) orow[t] = (bf16_t)0.f;
}

// ---------------- conv + SiLU, B channels ----------------
__global__ __launch_bounds__(128) void k_convB(const float* __restrict__ xbcdt,
                                               const float* __restrict__ conv_w,
                                               const float* __restrict__ conv_b,
                                               bf16_t* __restrict__ bw) {
  int idx = threadIdx.x;  // 8*16
  int b = idx >> 4, n = idx & 15;
  int c = DINNER + n;
  float w0 = conv_w[c * 4 + 0], w1 = conv_w[c * 4 + 1];
  float w2 = conv_w[c * 4 + 2], w3 = conv_w[c * 4 + 3];
  float bias = conv_b[c];
  const float* col = xbcdt + (size_t)b * TSEQ * NCOLS + c;
  bf16_t* orow = bw + (size_t)(b * DSTATE + n) * TPAD;
  float x0 = 0.f, x1 = 0.f, x2 = 0.f;
  for (int t = 0; t < TSEQ; ++t) {
    float x3 = col[(size_t)t * NCOLS];
    orow[t] = (bf16_t)siluf(x0 * w0 + x1 * w1 + x2 * w2 + x3 * w3 + bias);
    x0 = x1; x1 = x2; x2 = x3;
  }
  for (int t = TSEQ; t < TPAD; ++t) orow[t] = (bf16_t)0.f;
}

// ---------------- C at the last timestep only ----------------
__global__ __launch_bounds__(128) void k_clast(const float* __restrict__ xbcdt,
                                               const float* __restrict__ conv_w,
                                               const float* __restrict__ conv_b,
                                               float* __restrict__ clast) {
  int idx = threadIdx.x;  // 8*16
  int b = idx >> 4, n = idx & 15;
  int c = DINNER + DSTATE + n;
  float v = conv_b[c];
#pragma unroll
  for (int k = 0; k < 4; ++k) {
    int tt = (TSEQ - 1) - 3 + k;
    v += xbcdt[(size_t)(b * TSEQ + tt) * NCOLS + c] * conv_w[c * 4 + k];
  }
  clast[b * DSTATE + n] = siluf(v);
}

// ---- WMMA state reduction: h_T = Xw(64xT) @ Bw(Tx16), y = h_T @ C_last + D*x ----
__global__ __launch_bounds__(32) void k_scan(const bf16_t* __restrict__ xw,
                                             const bf16_t* __restrict__ bw,
                                             const float* __restrict__ clast,
                                             const float* __restrict__ xlast,
                                             const float* __restrict__ Dskip,
                                             float* __restrict__ ybuf) {
  int bh = blockIdx.x;
  int b = bh >> 5, h = bh & 31;
  int lane  = threadIdx.x;
  int khalf = lane >> 4, l15 = lane & 15;
  const bf16_t* ab = xw + (size_t)bh * HEADDIM * TPAD;
  const bf16_t* bb = bw + (size_t)b * DSTATE * TPAD;

  v8f acc[4];
#pragma unroll
  for (int mt = 0; mt < 4; ++mt) acc[mt] = zero8();

  for (int k0 = 0; k0 < TPAD; k0 += 32) {
    frag16 bf;
    const bf16_t* bp = bb + (size_t)l15 * TPAD + k0 + khalf * 8;
    bf.u[0] = *(const uint4*)(bp);
    bf.u[1] = *(const uint4*)(bp + 16);
#pragma unroll
    for (int mt = 0; mt < 4; ++mt) {
      frag16 af;
      const bf16_t* ap = ab + (size_t)(mt * 16 + l15) * TPAD + k0 + khalf * 8;
      af.u[0] = *(const uint4*)(ap);
      af.u[1] = *(const uint4*)(ap + 16);
      acc[mt] = wmma_bf16(af.v, bf.v, acc[mt]);
    }
  }

  float cn = clast[b * DSTATE + l15];
#pragma unroll
  for (int mt = 0; mt < 4; ++mt) {
#pragma unroll
    for (int j = 0; j < 8; ++j) {
      float v = acc[mt][j] * cn;           // contract over n (16 lanes per half)
      v += __shfl_xor(v, 1);
      v += __shfl_xor(v, 2);
      v += __shfl_xor(v, 4);
      v += __shfl_xor(v, 8);
      if (l15 == 0) {
        int p = mt * 16 + j + 8 * khalf;
        ybuf[(size_t)b * DINNER + h * HEADDIM + p] =
            v + Dskip[h] * xlast[bh * HEADDIM + p];
      }
    }
  }
}

// ---------------- z at last timestep: (8x2048) = x_t @ W[:2048]^T ----------------
__global__ __launch_bounds__(256) void k_z(const float* __restrict__ x_t,
                                           const float* __restrict__ in_proj_w,
                                           float* __restrict__ zbuf) {
  int idx = blockIdx.x * 256 + threadIdx.x;  // 8*2048
  int b = idx >> 11, e = idx & 2047;
  const float* wr = in_proj_w + (size_t)e * DMODEL;
  const float* xr = x_t + (size_t)b * DMODEL;
  float acc = 0.f;
  for (int d = 0; d < DMODEL; ++d) acc += wr[d] * xr[d];
  zbuf[idx] = acc;
}

// ---------------- gated RMSNorm ----------------
__global__ __launch_bounds__(256) void k_norm(const float* __restrict__ ybuf,
                                              const float* __restrict__ zbuf,
                                              const float* __restrict__ norm_w,
                                              float* __restrict__ yzn) {
  __shared__ float red[256];
  int b = blockIdx.x, tid = threadIdx.x;
  float loc = 0.f;
#pragma unroll
  for (int i = 0; i < 8; ++i) {
    int e = tid + i * 256;
    float y = ybuf[(size_t)b * DINNER + e];
    float z = zbuf[(size_t)b * DINNER + e];
    float yz = y * siluf(z);
    yzn[(size_t)b * DINNER + e] = yz;
    loc += yz * yz;
  }
  red[tid] = loc;
  __syncthreads();
  for (int s = 128; s > 0; s >>= 1) {
    if (tid < s) red[tid] += red[tid + s];
    __syncthreads();
  }
  float rms = rsqrtf(red[0] / (float)DINNER + 1e-5f);
#pragma unroll
  for (int i = 0; i < 8; ++i) {
    int e = tid + i * 256;
    yzn[(size_t)b * DINNER + e] *= rms * norm_w[e];
  }
}

// ---------------- out projection ----------------
__global__ __launch_bounds__(256) void k_out(const float* __restrict__ yzn,
                                             const float* __restrict__ out_proj_w,
                                             float* __restrict__ out) {
  int idx = blockIdx.x * 256 + threadIdx.x;  // 8*1024
  int b = idx >> 10, o = idx & 1023;
  const float* wr = out_proj_w + (size_t)o * DINNER;
  const float* yr = yzn + (size_t)b * DINNER;
  float acc = 0.f;
  for (int d = 0; d < DINNER; ++d) acc += wr[d] * yr[d];
  out[idx] = acc;
}

// ---------------- host launcher ----------------
extern "C" void kernel_launch(void* const* d_in, const int* in_sizes, int n_in,
                              void* d_out, int out_size, void* d_ws, size_t ws_size,
                              hipStream_t stream) {
  (void)in_sizes; (void)n_in; (void)out_size; (void)ws_size;
  const float* x_t        = (const float*)d_in[0];
  const float* history    = (const float*)d_in[1];
  const float* in_proj_w  = (const float*)d_in[2];
  const float* conv_w     = (const float*)d_in[3];
  const float* conv_b     = (const float*)d_in[4];
  const float* dt_bias    = (const float*)d_in[5];
  const float* A_log      = (const float*)d_in[6];
  const float* D_skip     = (const float*)d_in[7];
  const float* norm_w     = (const float*)d_in[8];
  const float* out_proj_w = (const float*)d_in[9];
  float* out = (float*)d_out;

  char* ws = (char*)d_ws;
  bf16_t* seqb  = (bf16_t*)ws;  ws += (size_t)MROWS * DMODEL * 2;
  bf16_t* w2b   = (bf16_t*)ws;  ws += (size_t)NCOLS * DMODEL * 2;
  float*  xbcdt = (float*)ws;   ws += (size_t)MROWS * NCOLS * 4;
  float*  wts   = (float*)ws;   ws += (size_t)BSZ * NHEADS * TPAD * 4;
  bf16_t* xw    = (bf16_t*)ws;  ws += (size_t)BSZ * NHEADS * HEADDIM * TPAD * 2;
  float*  xlast = (float*)ws;   ws += (size_t)BSZ * NHEADS * HEADDIM * 4;
  bf16_t* bw    = (bf16_t*)ws;  ws += (size_t)BSZ * DSTATE * TPAD * 2;
  float*  clast = (float*)ws;   ws += (size_t)BSZ * DSTATE * 4;
  float*  ybuf  = (float*)ws;   ws += (size_t)BSZ * DINNER * 4;
  float*  zbuf  = (float*)ws;   ws += (size_t)BSZ * DINNER * 4;
  float*  yzn   = (float*)ws;   ws += (size_t)BSZ * DINNER * 4;

  k_cast_seq<<<(MROWS * DMODEL) / 256, 256, 0, stream>>>(x_t, history, seqb);
  k_cast_w<<<(NCOLS * DMODEL) / 256, 256, 0, stream>>>(in_proj_w, w2b);
  k_gemm1<<<dim3((MROWS + BM - 1) / BM, NCOLS / BN), 256, 0, stream>>>(seqb, w2b, xbcdt);
  k_dt<<<BSZ * NHEADS, 32, 0, stream>>>(xbcdt, dt_bias, A_log, wts);
  k_convx<<<(BSZ * DINNER) / 256, 256, 0, stream>>>(xbcdt, conv_w, conv_b, wts, xw, xlast);
  k_convB<<<1, 128, 0, stream>>>(xbcdt, conv_w, conv_b, bw);
  k_clast<<<1, 128, 0, stream>>>(xbcdt, conv_w, conv_b, clast);
  k_z<<<(BSZ * DINNER) / 256, 256, 0, stream>>>(x_t, in_proj_w, zbuf);
  k_scan<<<BSZ * NHEADS, 32, 0, stream>>>(xw, bw, clast, xlast, D_skip, ybuf);
  k_norm<<<BSZ, 256, 0, stream>>>(ybuf, zbuf, norm_w, yzn);
  k_out<<<(BSZ * DMODEL) / 256, 256, 0, stream>>>(yzn, out_proj_w, out);
}
